// TransBlock_10290741641251
// MI455X (gfx1250) — matrix operand order repaired and checked by reading, not compiled, not run on hardware
//
#include <hip/hip_runtime.h>
#include <hip/hip_bf16.h>
#include <stdint.h>

#define X_DIM 1024
#define SEQ   1024
#define N3    3072

typedef __bf16 v8bf  __attribute__((ext_vector_type(8)));
typedef __bf16 v16bf __attribute__((ext_vector_type(16)));
typedef float  v8f   __attribute__((ext_vector_type(8)));
typedef float    f4  __attribute__((ext_vector_type(4)));
typedef unsigned u2  __attribute__((ext_vector_type(2)));
typedef unsigned u4  __attribute__((ext_vector_type(4)));
typedef unsigned u8  __attribute__((ext_vector_type(8)));

__device__ __forceinline__ unsigned short f2bf(float f) {
  union { float f; unsigned u; } v; v.f = f;
  return (unsigned short)((v.u + 0x7FFFu + ((v.u >> 16) & 1u)) >> 16);
}
__device__ __forceinline__ float bf2f(unsigned short s) {
  union { unsigned u; float f; } v; v.u = ((unsigned)s) << 16; return v.f;
}
__device__ __forceinline__ unsigned pack2bf(float a, float b) {
  return (unsigned)f2bf(a) | ((unsigned)f2bf(b) << 16);
}

// ---- CDNA5 Tensor Data Mover: 2-D bf16 tile (tile_d0 x tile_d1) global -> LDS ----
// D# per CDNA5 ISA. Inline asm form (toolchain-portable).
// Issued once per wave; tracked by TENSORcnt.
__device__ __forceinline__ void tdm_load_2d_bf16(unsigned lds_off, const void* gaddr,
                                                 unsigned tile_d0, unsigned tile_d1,
                                                 unsigned tensor_d0, unsigned tensor_d1,
                                                 unsigned stride_d0) {
  unsigned long long ga = (unsigned long long)gaddr;
  u4 g0;
  g0[0] = 1u;                                           // count=1, user mode, no gather
  g0[1] = lds_off;                                      // lds_addr [63:32]
  g0[2] = (unsigned)(ga & 0xFFFFFFFFu);                 // global_addr [95:64]
  g0[3] = (unsigned)((ga >> 32) & 0x1FFFFFFu) | (2u << 30);  // addr[56:32], type=2
  u8 g1;
  g1[0] = (1u << 16);                                   // wg_mask=0, data_size=1 (2B)
  g1[1] = (tensor_d0 & 0xFFFFu) << 16;                  // tensor_dim0[15:0] @ [63:48]
  g1[2] = ((tensor_d0 >> 16) & 0xFFFFu) | ((tensor_d1 & 0xFFFFu) << 16);
  g1[3] = ((tensor_d1 >> 16) & 0xFFFFu) | ((tile_d0 & 0xFFFFu) << 16);
  g1[4] = (tile_d1 & 0xFFFFu);                          // tile_dim1; tile_dim2=0 (2D)
  g1[5] = stride_d0;                                    // tensor_dim0_stride[31:0]
  g1[6] = 0;
  g1[7] = 0;
  asm volatile("tensor_load_to_lds %0, %1" :: "s"(g0), "s"(g1) : "memory");
}

// Async global->LDS 128-bit copy (GLOBAL_LOAD_ASYNC_TO_LDS_B128, ASYNCcnt).
__device__ __forceinline__ void async_b128(unsigned short* lds_dst, const unsigned short* gsrc) {
  unsigned off = (unsigned)(uintptr_t)lds_dst;
  asm volatile("global_load_async_to_lds_b128 %0, %1, off"
               :: "v"(off), "v"(gsrc) : "memory");
}
__device__ __forceinline__ void wait_async0(void) {
  asm volatile("s_wait_asynccnt 0x0" ::: "memory");
}

// A fragment (16x32 bf16, MxK): lane L holds row L%16; K chunk base 0 (lanes 0-15)
// or 8 (lanes 16-31); VGPRs 0-3 hold K=[kb..kb+7], VGPRs 4-7 hold K=[kb+16..kb+23].
__device__ __forceinline__ v16bf frag_A(const unsigned short* t, int row, int lane) {
  const int kb = (lane & 16) ? 8 : 0;
  v8bf lo = *reinterpret_cast<const v8bf*>(t + row * 32 + kb);
  v8bf hi = *reinterpret_cast<const v8bf*>(t + row * 32 + kb + 16);
  return __builtin_shufflevector(lo, hi, 0,1,2,3,4,5,6,7,8,9,10,11,12,13,14,15);
}
// B fragment (32x16 bf16, KxN): lane L holds col L%16; K chunk base 0 (lanes 0-15)
// or 8 (lanes 16-31); VGPRs 0-3 hold K=[kb..kb+7], VGPRs 4-7 hold K=[kb+16..kb+23]. LDS tile is [col][k].
__device__ __forceinline__ v16bf frag_B(const unsigned short* t, int col, int lane) {
  const int kb = (lane & 16) ? 8 : 0;
  v8bf lo = *reinterpret_cast<const v8bf*>(t + col * 32 + kb);
  v8bf hi = *reinterpret_cast<const v8bf*>(t + col * 32 + kb + 16);
  return __builtin_shufflevector(lo, hi, 0,1,2,3,4,5,6,7,8,9,10,11,12,13,14,15);
}

// One K=32 step: wave (wm,wn) computes a 32x64 patch of the 128x128 block tile.
__device__ __forceinline__ void mma_step(const unsigned short* la, const unsigned short* lb,
                                         v8f acc[2][4], int wm, int wn, int lane) {
  v16bf a[2], b[4];
#pragma unroll
  for (int mt = 0; mt < 2; ++mt) a[mt] = frag_A(la, wm * 32 + mt * 16 + (lane & 15), lane);
#pragma unroll
  for (int nt = 0; nt < 4; ++nt) b[nt] = frag_B(lb, wn * 64 + nt * 16 + (lane & 15), lane);
#pragma unroll
  for (int mt = 0; mt < 2; ++mt)
#pragma unroll
    for (int nt = 0; nt < 4; ++nt)
      acc[mt][nt] = __builtin_amdgcn_wmma_f32_16x16x32_bf16(
          false, a[mt], false, b[nt], (short)0, acc[mt][nt], false, false);
}

// ---------------- Kernel 1: qkv = x @ W + b  (fp32 in -> bf16 out) ----------------
__global__ __launch_bounds__(256) void qkv_gemm_kernel(
    const float* __restrict__ x, const float* __restrict__ W,
    const float* __restrict__ bias, unsigned short* __restrict__ qkv) {
  __shared__ __align__(16) unsigned short la[128 * 32];
  __shared__ __align__(16) unsigned short lb[128 * 32];
  const int tid = threadIdx.x, lane = tid & 31, wv = tid >> 5;
  const int wm = wv & 3, wn = wv >> 2;
  const int n0 = blockIdx.x * 128, s0 = blockIdx.y * 128, b = blockIdx.z;
  const size_t xbase = ((size_t)b * SEQ + s0) * X_DIM;
  const size_t qbase = (size_t)b * SEQ * N3;
  const v8f vz = {0.f, 0.f, 0.f, 0.f, 0.f, 0.f, 0.f, 0.f};
  v8f acc[2][4];
  for (int mt = 0; mt < 2; ++mt)
    for (int nt = 0; nt < 4; ++nt) acc[mt][nt] = vz;

  for (int kt = 0; kt < X_DIM / 32; ++kt) {
    const int k0 = kt * 32;
#pragma unroll
    for (int i = 0; i < 4; ++i) {                     // A tile: b128 load, b64 store
      int idx = i * 1024 + tid * 4;
      int row = idx >> 5, k = idx & 31;
      f4 xv = *reinterpret_cast<const f4*>(&x[xbase + (size_t)row * X_DIM + k0 + k]);
      u2 p; p.x = pack2bf(xv.x, xv.y); p.y = pack2bf(xv.z, xv.w);
      *reinterpret_cast<u2*>(&la[idx]) = p;
    }
#pragma unroll
    for (int i = 0; i < 2; ++i) {                     // W^T tile: 2 rows -> packed b32
      int j = i * 256 + tid;                          // 0..511
      int n = (j & 31) * 4, kk = (j >> 5) * 2;        // n mult of 4, kk even
      f4 w0 = *reinterpret_cast<const f4*>(&W[(size_t)(k0 + kk) * N3 + n0 + n]);
      f4 w1 = *reinterpret_cast<const f4*>(&W[(size_t)(k0 + kk + 1) * N3 + n0 + n]);
      *reinterpret_cast<unsigned*>(&lb[(n + 0) * 32 + kk]) = pack2bf(w0.x, w1.x);
      *reinterpret_cast<unsigned*>(&lb[(n + 1) * 32 + kk]) = pack2bf(w0.y, w1.y);
      *reinterpret_cast<unsigned*>(&lb[(n + 2) * 32 + kk]) = pack2bf(w0.z, w1.z);
      *reinterpret_cast<unsigned*>(&lb[(n + 3) * 32 + kk]) = pack2bf(w0.w, w1.w);
    }
    if (kt + 1 < X_DIM / 32)
      __builtin_prefetch(&x[xbase + (size_t)(tid >> 1) * X_DIM + k0 + 32], 0, 1);
    __syncthreads();
    mma_step(la, lb, acc, wm, wn, lane);
    __syncthreads();
  }
#pragma unroll
  for (int mt = 0; mt < 2; ++mt)
#pragma unroll
    for (int nt = 0; nt < 4; ++nt)
#pragma unroll
      for (int r = 0; r < 8; ++r) {
        int row = s0 + wm * 32 + mt * 16 + r + ((lane >> 4) << 3);
        int col = n0 + wn * 64 + nt * 16 + (lane & 15);
        qkv[qbase + (size_t)row * N3 + col] = f2bf(acc[mt][nt][r] + bias[col]);
      }
}

// ---- Kernel 2: scores = scale * Q @ K^T (bf16, TDM double-buffered tiles) ----
__global__ __launch_bounds__(256) void scores_kernel(
    const unsigned short* __restrict__ qkv, unsigned short* __restrict__ scores) {
  __shared__ __align__(16) unsigned short la[2][128 * 32];
  __shared__ __align__(16) unsigned short lb[2][128 * 32];
  const int tid = threadIdx.x, lane = tid & 31, wv = tid >> 5;
  const int wm = wv & 3, wn = wv >> 2;
  const int kq0 = blockIdx.x * 128, q0 = blockIdx.y * 128, b = blockIdx.z;
  const size_t qb = (size_t)b * SEQ * N3;
  const size_t sb = (size_t)b * SEQ * SEQ;
  const float scale = 0.03125f;  // 1/sqrt(1024)
  const v8f vz = {0.f, 0.f, 0.f, 0.f, 0.f, 0.f, 0.f, 0.f};
  v8f acc[2][4];
  for (int mt = 0; mt < 2; ++mt)
    for (int nt = 0; nt < 4; ++nt) acc[mt][nt] = vz;

  // Wave 0 issues one TDM per operand tile: 128 rows x 32 bf16, row stride 3072.
  auto issue = [&](int kt, int s) {
    const int d0 = kt * 32;
    tdm_load_2d_bf16((unsigned)(uintptr_t)&la[s][0],
                     &qkv[qb + (size_t)q0 * N3 + d0],            // Q tile start
                     32, 128, N3, SEQ, N3);
    tdm_load_2d_bf16((unsigned)(uintptr_t)&lb[s][0],
                     &qkv[qb + (size_t)kq0 * N3 + X_DIM + d0],   // K tile start
                     32, 128, N3, SEQ, N3);
  };

  if (wv == 0) issue(0, 0);
  for (int kt = 0; kt < X_DIM / 32; ++kt) {
    const int cur = kt & 1;
    const bool more = (kt + 1 < X_DIM / 32);
    if (wv == 0) {
      if (more) issue(kt + 1, cur ^ 1);
      if (more) __builtin_amdgcn_s_wait_tensorcnt(2);  // tile kt done, kt+1 in flight
      else      __builtin_amdgcn_s_wait_tensorcnt(0);
    }
    __syncthreads();
    mma_step(la[cur], lb[cur], acc, wm, wn, lane);
    __syncthreads();
  }
#pragma unroll
  for (int mt = 0; mt < 2; ++mt)
#pragma unroll
    for (int nt = 0; nt < 4; ++nt)
#pragma unroll
      for (int r = 0; r < 8; ++r) {
        int row = q0 + wm * 32 + mt * 16 + r + ((lane >> 4) << 3);
        int col = kq0 + wn * 64 + nt * 16 + (lane & 15);
        scores[sb + (size_t)row * SEQ + col] = f2bf(acc[mt][nt][r] * scale);
      }
}

// ---------------- Kernel 3: row softmax in place (bf16) ----------------
__global__ __launch_bounds__(256) void softmax_kernel(unsigned short* __restrict__ scores) {
  __shared__ float red[256];
  const int tid = threadIdx.x;
  const size_t base = (size_t)blockIdx.x * SEQ;
  float v[4];
  float m = -3.4e38f;
#pragma unroll
  for (int i = 0; i < 4; ++i) {
    v[i] = bf2f(scores[base + tid * 4 + i]);
    m = fmaxf(m, v[i]);
  }
  red[tid] = m;
  __syncthreads();
  for (int s = 128; s > 0; s >>= 1) {
    if (tid < s) red[tid] = fmaxf(red[tid], red[tid + s]);
    __syncthreads();
  }
  m = red[0];
  __syncthreads();
  float sum = 0.f;
#pragma unroll
  for (int i = 0; i < 4; ++i) {
    v[i] = __expf(v[i] - m);
    sum += v[i];
  }
  red[tid] = sum;
  __syncthreads();
  for (int s = 128; s > 0; s >>= 1) {
    if (tid < s) red[tid] += red[tid + s];
    __syncthreads();
  }
  const float inv = 1.0f / red[0];
#pragma unroll
  for (int i = 0; i < 4; ++i)
    scores[base + tid * 4 + i] = f2bf(v[i] * inv);
}

// ---------------- Kernel 4: out = P @ V  (bf16 -> fp32, async A tile) ----------------
__global__ __launch_bounds__(256) void out_gemm_kernel(
    const unsigned short* __restrict__ qkv, const unsigned short* __restrict__ probs,
    float* __restrict__ out) {
  __shared__ __align__(16) unsigned short la[128 * 32];
  __shared__ __align__(16) unsigned short lb[128 * 32];
  const int tid = threadIdx.x, lane = tid & 31, wv = tid >> 5;
  const int wm = wv & 3, wn = wv >> 2;
  const int d0 = blockIdx.x * 128, q0 = blockIdx.y * 128, b = blockIdx.z;
  const size_t qb = (size_t)b * SEQ * N3;
  const size_t sb = (size_t)b * SEQ * SEQ;
  const size_t ob = (size_t)b * SEQ * X_DIM;
  const v8f vz = {0.f, 0.f, 0.f, 0.f, 0.f, 0.f, 0.f, 0.f};
  v8f acc[2][4];
  for (int mt = 0; mt < 2; ++mt)
    for (int nt = 0; nt < 4; ++nt) acc[mt][nt] = vz;

  for (int kt = 0; kt < SEQ / 32; ++kt) {
    const int k0 = kt * 32;
#pragma unroll
    for (int i = 0; i < 2; ++i) {                  // P tile: async B128 copy
      int idx = i * 2048 + tid * 8;
      int row = idx >> 5, k = idx & 31;
      async_b128(&la[idx], &probs[sb + (size_t)(q0 + row) * SEQ + k0 + k]);
    }
#pragma unroll
    for (int i = 0; i < 2; ++i) {                  // V^T tile: b128 load, strided ds store
      int idx = i * 2048 + tid * 8;
      int n = idx & 127, kk = idx >> 7;            // n multiple of 8
      u4 vv = *reinterpret_cast<const u4*>(&qkv[qb + (size_t)(k0 + kk) * N3 + 2 * X_DIM + d0 + n]);
      const unsigned short* h = reinterpret_cast<const unsigned short*>(&vv);
#pragma unroll
      for (int j = 0; j < 8; ++j) lb[(n + j) * 32 + kk] = h[j];
    }
    wait_async0();
    __syncthreads();
    mma_step(la, lb, acc, wm, wn, lane);
    __syncthreads();
  }
#pragma unroll
  for (int mt = 0; mt < 2; ++mt)
#pragma unroll
    for (int nt = 0; nt < 4; ++nt)
#pragma unroll
      for (int r = 0; r < 8; ++r) {
        int row = q0 + wm * 32 + mt * 16 + r + ((lane >> 4) << 3);
        int col = d0 + wn * 64 + nt * 16 + (lane & 15);
        out[ob + (size_t)row * X_DIM + col] = acc[mt][nt][r];
      }
}

extern "C" void kernel_launch(void* const* d_in, const int* in_sizes, int n_in,
                              void* d_out, int out_size, void* d_ws, size_t ws_size,
                              hipStream_t stream) {
  const float* x = (const float*)d_in[0];      // [32,1024,1024]
  const float* W = (const float*)d_in[1];      // [1024,3072]
  const float* bias = (const float*)d_in[2];   // [3072]
  float* out = (float*)d_out;                  // [32,1024,1024]

  const int B = 32;
  unsigned short* qkv = (unsigned short*)d_ws;                 // B*S*3072 bf16
  unsigned short* scores = qkv + (size_t)B * SEQ * N3;         // B*S*S bf16

  qkv_gemm_kernel<<<dim3(N3 / 128, SEQ / 128, B), 256, 0, stream>>>(x, W, bias, qkv);
  scores_kernel<<<dim3(SEQ / 128, SEQ / 128, B), 256, 0, stream>>>(qkv, scores);
  softmax_kernel<<<dim3(B * SEQ), 256, 0, stream>>>(scores);
  out_gemm_kernel<<<dim3(X_DIM / 128, SEQ / 128, B), 256, 0, stream>>>(qkv, scores, out);
}
